// LSTM_28810640622288
// MI455X (gfx1250) — compile-verified, matrix operand order checked by reading, not run on hardware
//
#include <hip/hip_runtime.h>
#include <math.h>

// ---------------- problem constants ----------------
#define EMBED   768
#define HIDDEN  1024
#define G4      (4 * HIDDEN)      // 4096 gate rows
#define TDIM    512
#define BDIM    64
#define TAGS    32

// recurrence partitioning: 128 WGs per layer, each owns 8 h-dims (32 gate rows)
#define NWG_L       128
#define NWG_TOT     (2 * NWG_L)
#define JD          (HIDDEN / NWG_L)     // 8 h-dims per WG
#define ROWS_WG     (4 * JD)             // 32 gate rows per WG
#define SLAB_BYTES  (ROWS_WG * HIDDEN * 2)   // 64 KB per weight slab

typedef __attribute__((ext_vector_type(16))) __bf16 v16bf;
typedef __attribute__((ext_vector_type(8)))  __bf16 v8bf;
typedef __attribute__((ext_vector_type(8)))  float  v8f;

// =====================================================================
// Prep kernels: bf16 conversion / gathers / packing
// =====================================================================
__global__ void k_zero(unsigned* sync, __bf16* h1s, __bf16* h2s) {
    int t = threadIdx.x;
    if (t < 4) sync[t] = 0u;
    if (t < HIDDEN) { h1s[t] = (__bf16)0.0f; h2s[t] = (__bf16)0.0f; } // t = -1 state
}

__global__ void k_f2bf(const float* __restrict__ s, __bf16* __restrict__ d, int n) {
    int i = blockIdx.x * blockDim.x + threadIdx.x;
    if (i < n) d[i] = (__bf16)s[i];
}

// extract batch element 63 of x: (T, B, E) -> (T, E) bf16
__global__ void k_gather_x63(const float* __restrict__ x, __bf16* __restrict__ d) {
    int i = blockIdx.x * blockDim.x + threadIdx.x;
    if (i < TDIM * EMBED) {
        int t = i / EMBED, e = i % EMBED;
        d[i] = (__bf16)x[((size_t)t * BDIM + (BDIM - 1)) * EMBED + e];
    }
}

__global__ void k_bias(const float* __restrict__ a, const float* __restrict__ b,
                       float* __restrict__ o, int n) {
    int i = blockIdx.x * blockDim.x + threadIdx.x;
    if (i < n) o[i] = a[i] + b[i];
}

// pack a (4096 x 1024) weight into per-WG LDS-friendly slabs:
// dst[wg][lr][k], lr = gate*JD + jl  <->  src row = gate*HIDDEN + wg*JD + jl
__global__ void k_pack(const float* __restrict__ W, __bf16* __restrict__ d, int n) {
    int i = blockIdx.x * blockDim.x + threadIdx.x;
    if (i < n) {
        int k  = i & (HIDDEN - 1);
        int lr = (i >> 10) & (ROWS_WG - 1);
        int wg = i >> 15;
        int gate = lr / JD, jl = lr % JD;
        int grow = gate * HIDDEN + wg * JD + jl;
        d[i] = (__bf16)W[(size_t)grow * HIDDEN + k];
    }
}

// =====================================================================
// WMMA bf16 GEMM:  C(MxN) = A(MxK) * B(NxK)^T + bias[n]
// One wave per 16x16 output tile; fragments per ISA 16-bit A-layout.
// =====================================================================
__device__ __forceinline__ v16bf load_frag(const __bf16* __restrict__ base,
                                           int ld, int row, int kb, int lane) {
    int l  = lane & 15;
    int hi = lane >> 4;                     // 0: K 0-7/16-23, 1: K 8-15/24-31
    const __bf16* p = base + (size_t)(row + l) * ld + kb + hi * 8;
    v8bf a = *(const v8bf*)p;
    v8bf b = *(const v8bf*)(p + 16);
    v16bf r;
    #pragma unroll
    for (int i = 0; i < 8; ++i) { r[i] = a[i]; r[i + 8] = b[i]; }
    return r;
}

__global__ void __launch_bounds__(256)
k_wmma_gemm(const __bf16* __restrict__ A, const __bf16* __restrict__ B,
            const float* __restrict__ bias, float* __restrict__ C,
            int M, int N, int K) {
    int wid  = (blockIdx.x * blockDim.x + threadIdx.x) >> 5;
    int lane = threadIdx.x & 31;
    int tilesN = N >> 4;
    int tM = wid / tilesN;
    int tN = wid - tM * tilesN;
    if (tM * 16 >= M) return;               // wave-uniform guard (EXEC all-1 at wmma)

    v8f acc = {0.f, 0.f, 0.f, 0.f, 0.f, 0.f, 0.f, 0.f};
    for (int kb = 0; kb < K; kb += 32) {
        v16bf af = load_frag(A, K, tM * 16, kb, lane);
        v16bf bf = load_frag(B, K, tN * 16, kb, lane);
        acc = __builtin_amdgcn_wmma_f32_16x16x32_bf16(
                  false, af, false, bf, (short)0, acc, false, false);
    }
    // C/D layout: lanes 0-15: N=lane, M=r; lanes 16-31: N=lane-16, M=r+8
    int col   = tN * 16 + (lane & 15);
    int rbase = tM * 16 + ((lane >> 4) << 3);
    float bv  = bias ? bias[col] : 0.f;
    #pragma unroll
    for (int r = 0; r < 8; ++r)
        C[(size_t)(rbase + r) * N + col] = acc[r] + bv;
}

// =====================================================================
// Tensor Data Mover: 1-D tile DMA, global -> LDS (gfx1250 TDM, D# per ISA §8)
// 6-arg builtin variant (clang-23 / therock-10.0 headers).
// =====================================================================
#if __has_builtin(__builtin_amdgcn_tensor_load_to_lds)
#define HAVE_TDM 1
typedef __attribute__((ext_vector_type(4))) unsigned tdm_u4;
typedef __attribute__((ext_vector_type(8))) int      tdm_i8;
typedef __attribute__((ext_vector_type(4))) int      tdm_i4;

__device__ __forceinline__ void tdm_load_1d(const void* gptr, unsigned lds_off,
                                            unsigned bytes) {
    const unsigned n8 = bytes >> 3;                 // 8-byte elements
    const unsigned long long ga = (unsigned long long)(uintptr_t)gptr;
    tdm_u4 g0;
    g0[0] = 1u;                                     // count=1, user mode
    g0[1] = lds_off;                                // lds_addr (byte offset)
    g0[2] = (unsigned)(ga & 0xffffffffu);           // global_addr[31:0]
    g0[3] = (unsigned)((ga >> 32) & 0x1ffffffu)     // global_addr[56:32]
          | (2u << 30);                             // type=2 ("image")
    tdm_i8 g1;
    g1[0] = (int)(3u << 16);                        // data_size=8B; no multicast
    g1[1] = (int)((n8 & 0xffffu) << 16);            // tensor_dim0[15:0]
    g1[2] = (int)((n8 >> 16) & 0xffffu);            // tensor_dim0[31:16]; dim1 unused
    g1[3] = (int)((n8 & 0xffffu) << 16);            // tile_dim0 (1-D tile)
    g1[4] = 0; g1[5] = 0; g1[6] = 0; g1[7] = 0;     // tile_dim1/2=0, strides unused
    tdm_i4 z4 = {0, 0, 0, 0};
    tdm_i8 z8 = {0, 0, 0, 0, 0, 0, 0, 0};
    __builtin_amdgcn_tensor_load_to_lds(g0, g1, z4, z4, z8, 0);
}
#else
#define HAVE_TDM 0
#endif

// =====================================================================
// Persistent pipelined 2-layer LSTM recurrence (batch element 63 only).
// grid = 256 WGs: [0,128) layer0, [128,256) layer1 (pipelined one step behind).
// Weights resident in LDS (up to 128KB/WGP), staged via TDM. Grid barrier/step.
// =====================================================================
__device__ __forceinline__ void grid_sync(volatile unsigned* cnt2, volatile unsigned* gen,
                                          unsigned nwg, unsigned target) {
    __syncthreads();
    if (threadIdx.x == 0) {
        __threadfence();
        volatile unsigned* cnt = &cnt2[target & 1];
        unsigned prev = atomicAdd((unsigned*)cnt, 1u);
        if (prev == nwg - 1) {
            *cnt = 0;                       // reset before releasing: sense-safe
            __threadfence();
            atomicAdd((unsigned*)gen, 1u);
        } else {
            while (*gen < target) __builtin_amdgcn_s_sleep(2);
        }
    }
    __syncthreads();
}

__global__ void __launch_bounds__(256, 1)
k_rec(const __bf16* __restrict__ pack0, const __bf16* __restrict__ packI1,
      const __bf16* __restrict__ packH1, const float* __restrict__ xg0,
      const float* __restrict__ bias1, __bf16* __restrict__ h1s,
      __bf16* __restrict__ h2s, unsigned* sync) {
    extern __shared__ char smem_[];
    __bf16* wA    = (__bf16*)smem_;                       // 32x1024 bf16
    __bf16* wB    = wA + ROWS_WG * HIDDEN;                // layer1 only
    float*  hbufA = (float*)(wB + ROWS_WG * HIDDEN);      // 1024 f32
    float*  hbufB = hbufA + HIDDEN;                       // 1024 f32
    float*  gbuf  = hbufB + HIDDEN;                       // 32
    float*  cbuf  = gbuf + ROWS_WG;                       // 8 (cell state)

    const int  tid = threadIdx.x;
    const bool L1  = blockIdx.x >= NWG_L;
    const int  lw  = L1 ? blockIdx.x - NWG_L : blockIdx.x;

    // stage resident weights into LDS
    const __bf16* slabA = (L1 ? packI1 : pack0) + (size_t)lw * ROWS_WG * HIDDEN;
    const __bf16* slabB = packH1 + (size_t)lw * ROWS_WG * HIDDEN;
#if HAVE_TDM
    if (tid < 32) {                                       // wave 0 issues the DMA
        const unsigned dynBase = __builtin_amdgcn_groupstaticsize();
        tdm_load_1d(slabA, dynBase + (unsigned)((char*)wA - smem_), SLAB_BYTES);
        if (L1)
            tdm_load_1d(slabB, dynBase + (unsigned)((char*)wB - smem_), SLAB_BYTES);
        __builtin_amdgcn_s_wait_tensorcnt(0);
    }
#else
    {
        const uint4* s0 = (const uint4*)slabA;
        uint4* dA = (uint4*)wA;
        const int n16 = SLAB_BYTES / 16;
        for (int i = tid; i < n16; i += 256) dA[i] = s0[i];
        if (L1) {
            const uint4* s1 = (const uint4*)slabB;
            uint4* dB = (uint4*)wB;
            for (int i = tid; i < n16; i += 256) dB[i] = s1[i];
        }
    }
#endif
    if (tid < JD) cbuf[tid] = 0.f;
    __syncthreads();

    unsigned target = 0;
    for (int s = 0; s <= TDIM; ++s) {
        const bool active = L1 ? (s >= 1) : (s < TDIM);
        const int  t = L1 ? s - 1 : s;
        if (active) {
            // fetch h vectors (bf16 global -> f32 LDS). Slot convention: slot t+1 == h[t].
            if (!L1) {
                for (int i = tid; i < HIDDEN; i += 256)
                    hbufA[i] = (float)h1s[(size_t)t * HIDDEN + i];            // h1[t-1]
            } else {
                for (int i = tid; i < HIDDEN; i += 256) {
                    hbufA[i] = (float)h1s[(size_t)(t + 1) * HIDDEN + i];     // h1[t] (input)
                    hbufB[i] = (float)h2s[(size_t)t * HIDDEN + i];           // h2[t-1]
                }
            }
            __syncthreads();

            // 32 gate rows x 1024 dot; 8 lanes per row; bf16 pairs per ds_load_b32
            const int lr  = tid >> 3;
            const int sub = tid & 7;
            float acc = 0.f;
            {
                const unsigned* wr = (const unsigned*)(wA + (size_t)lr * HIDDEN) + sub * 64;
                const float2*   hp = (const float2*)hbufA + sub * 64;
                #pragma unroll 8
                for (int k = 0; k < 64; ++k) {
                    unsigned w2 = wr[k];
                    float2 hv = hp[k];
                    acc = fmaf(__uint_as_float(w2 << 16),         hv.x, acc);
                    acc = fmaf(__uint_as_float(w2 & 0xffff0000u), hv.y, acc);
                }
            }
            if (L1) {
                const unsigned* wr = (const unsigned*)(wB + (size_t)lr * HIDDEN) + sub * 64;
                const float2*   hp = (const float2*)hbufB + sub * 64;
                #pragma unroll 8
                for (int k = 0; k < 64; ++k) {
                    unsigned w2 = wr[k];
                    float2 hv = hp[k];
                    acc = fmaf(__uint_as_float(w2 << 16),         hv.x, acc);
                    acc = fmaf(__uint_as_float(w2 & 0xffff0000u), hv.y, acc);
                }
            }
            acc += __shfl_xor(acc, 1, 32);
            acc += __shfl_xor(acc, 2, 32);
            acc += __shfl_xor(acc, 4, 32);
            if (sub == 0) {
                int gate = lr / JD, jl = lr % JD;
                int grow = gate * HIDDEN + lw * JD + jl;
                float base = L1 ? bias1[grow] : xg0[(size_t)t * G4 + grow];
                gbuf[lr] = acc + base;                    // gate order: i,f,g,o banks
                // hide next step's xg0 L2 fetch behind the grid barrier
                if (!L1 && t + 1 < TDIM)
                    __builtin_prefetch(&xg0[(size_t)(t + 1) * G4 + grow], 0, 1);
            }
            __syncthreads();

            if (tid < JD) {
                float gi = gbuf[tid];
                float gf = gbuf[JD + tid];
                float gg = gbuf[2 * JD + tid];
                float go = gbuf[3 * JD + tid];
                float iv = 1.f / (1.f + expf(-gi));
                float fv = 1.f / (1.f + expf(-gf));
                float gv = tanhf(gg);
                float ov = 1.f / (1.f + expf(-go));
                float c  = fv * cbuf[tid] + iv * gv;
                cbuf[tid] = c;
                float h  = ov * tanhf(c);
                int j = lw * JD + tid;
                if (!L1) h1s[(size_t)(t + 1) * HIDDEN + j] = (__bf16)h;
                else     h2s[(size_t)(t + 1) * HIDDEN + j] = (__bf16)h;
            }
        }
        ++target;
        grid_sync(sync, sync + 2, NWG_TOT, target);
    }
}

// =====================================================================
// log_softmax over 32 tags, one wave per timestep row
// =====================================================================
__global__ void k_lsm(const float* __restrict__ tagv, float* __restrict__ out) {
    int t = blockIdx.x, l = threadIdx.x;
    float v = tagv[t * TAGS + l];
    float m = v;
    #pragma unroll
    for (int o = 16; o >= 1; o >>= 1) m = fmaxf(m, __shfl_xor(m, o, 32));
    float e = expf(v - m), ssum = e;
    #pragma unroll
    for (int o = 16; o >= 1; o >>= 1) ssum += __shfl_xor(ssum, o, 32);
    out[t * TAGS + l] = (v - m) - logf(ssum);
}

// =====================================================================
// host launcher
// =====================================================================
extern "C" void kernel_launch(void* const* d_in, const int* in_sizes, int n_in,
                              void* d_out, int out_size, void* d_ws, size_t ws_size,
                              hipStream_t stream) {
    const float* x     = (const float*)d_in[0];
    const float* Wih0  = (const float*)d_in[1];
    const float* Whh0  = (const float*)d_in[2];
    const float* bih0  = (const float*)d_in[3];
    const float* bhh0  = (const float*)d_in[4];
    const float* Wih1  = (const float*)d_in[5];
    const float* Whh1  = (const float*)d_in[6];
    const float* bih1  = (const float*)d_in[7];
    const float* bhh1  = (const float*)d_in[8];
    const float* Wtag  = (const float*)d_in[9];
    const float* btag  = (const float*)d_in[10];
    float* out = (float*)d_out;

    char*  ws  = (char*)d_ws;
    size_t off = 0;
    auto carve = [&](size_t bytes) -> char* {
        char* p = ws + off;
        off = (off + bytes + 255) & ~(size_t)255;
        return p;
    };
    unsigned* syncp   = (unsigned*)carve(4 * sizeof(unsigned));
    __bf16*   x63     = (__bf16*)carve((size_t)TDIM * EMBED * 2);
    __bf16*   wih0b   = (__bf16*)carve((size_t)G4 * EMBED * 2);
    __bf16*   wtagb   = (__bf16*)carve((size_t)TAGS * HIDDEN * 2);
    float*    bias0   = (float*)carve((size_t)G4 * 4);
    float*    bias1   = (float*)carve((size_t)G4 * 4);
    __bf16*   pack0   = (__bf16*)carve((size_t)NWG_L * SLAB_BYTES);
    __bf16*   packI1  = (__bf16*)carve((size_t)NWG_L * SLAB_BYTES);
    __bf16*   packH1  = (__bf16*)carve((size_t)NWG_L * SLAB_BYTES);
    float*    xg0     = (float*)carve((size_t)TDIM * G4 * 4);
    __bf16*   h1s     = (__bf16*)carve((size_t)(TDIM + 1) * HIDDEN * 2);
    __bf16*   h2s     = (__bf16*)carve((size_t)(TDIM + 1) * HIDDEN * 2);
    float*    tagbuf  = (float*)carve((size_t)TDIM * TAGS * 4);

    auto blk = [](int n) { return (n + 255) / 256; };

    // prep
    k_zero<<<1, 1024, 0, stream>>>(syncp, h1s, h2s);
    k_gather_x63<<<blk(TDIM * EMBED), 256, 0, stream>>>(x, x63);
    k_f2bf<<<blk(G4 * EMBED), 256, 0, stream>>>(Wih0, wih0b, G4 * EMBED);
    k_f2bf<<<blk(TAGS * HIDDEN), 256, 0, stream>>>(Wtag, wtagb, TAGS * HIDDEN);
    k_bias<<<blk(G4), 256, 0, stream>>>(bih0, bhh0, bias0, G4);
    k_bias<<<blk(G4), 256, 0, stream>>>(bih1, bhh1, bias1, G4);
    const int npack = NWG_L * ROWS_WG * HIDDEN;
    k_pack<<<blk(npack), 256, 0, stream>>>(Whh0, pack0, npack);
    k_pack<<<blk(npack), 256, 0, stream>>>(Wih1, packI1, npack);
    k_pack<<<blk(npack), 256, 0, stream>>>(Whh1, packH1, npack);

    // xg0 = x63 @ W_ih0^T + (b_ih0 + b_hh0)   [WMMA bf16, f32 accum]
    {
        int waves = (TDIM / 16) * (G4 / 16);      // 8192
        k_wmma_gemm<<<waves * 32 / 256, 256, 0, stream>>>(
            x63, wih0b, bias0, xg0, TDIM, G4, EMBED);
    }

    // pipelined persistent 2-layer recurrence
    {
        size_t smem = (size_t)2 * ROWS_WG * HIDDEN * 2   // wA + wB
                    + (size_t)2 * HIDDEN * 4             // hbufA + hbufB
                    + ROWS_WG * 4 + JD * 4;              // gbuf + cbuf
        k_rec<<<NWG_TOT, 256, smem, stream>>>(
            pack0, packI1, packH1, xg0, bias1, h1s, h2s, syncp);
    }

    // tag head: (512 x 1024) @ (1024 x 32) + b_tag   [WMMA]
    {
        int waves = (TDIM / 16) * (TAGS / 16);    // 64
        k_wmma_gemm<<<waves * 32 / 256, 256, 0, stream>>>(
            h2s + HIDDEN, wtagb, btag, tagbuf, TDIM, TAGS, HIDDEN);
    }

    // log_softmax
    k_lsm<<<TDIM, TAGS, 0, stream>>>(tagbuf, out);
}
